// TokenMerger_30666066494182
// MI455X (gfx1250) — matrix operand order repaired and checked by reading, not compiled, not run on hardware
//
#include <hip/hip_runtime.h>
#include <hip/hip_bf16.h>

// ---------------------------------------------------------------------------
// Token merger (ToMe-style) for MI455X / gfx1250 (wave32, WMMA, 320KB LDS).
// Heavy phase: fused similarity GEMM (a @ b^T, scales applied at epilogue)
// + row max/argmax, via v_wmma_f32_16x16x32_f16. The 128-row A tile stays
// resident in LDS for the whole block (275KB dynamic LDS - CDNA5 only).
// ---------------------------------------------------------------------------

typedef _Float16 v16h __attribute__((ext_vector_type(16)));
typedef _Float16 v8h  __attribute__((ext_vector_type(8)));
typedef float    v8f  __attribute__((ext_vector_type(8)));
typedef float    v4f  __attribute__((ext_vector_type(4)));

#define B_   8
#define T_   4096
#define D_   1024
#define N_   2048          // n = T/2
#define R_   1024          // r = T * 0.25
#define TILE_M 128         // A rows resident in LDS per block
#define TILE_N 64          // B rows (columns of scores) per N-tile
#define KCH    128         // K chunk of B resident in LDS
#define LDA    1032        // A LDS row stride in halves (1024+8: bank-conflict-free)
#define LDB    136         // B LDS row stride in halves (128+8)

#define SMEM_BYTES ((TILE_M * LDA + TILE_N * LDB) * sizeof(_Float16))  // 281,600 B

// ---------------------------------------------------------------------------
// Kernel 1: per-token-row inverse L2 norm (clamped at 1e-12 like F.normalize)
// grid = B*T blocks, 128 threads
// ---------------------------------------------------------------------------
__global__ __launch_bounds__(128) void norm_kernel(const float* __restrict__ x,
                                                   float* __restrict__ inv_na,
                                                   float* __restrict__ inv_nb) {
  const int rid = blockIdx.x;                 // token row 0..B*T-1
  const int tid = threadIdx.x;
  const float* row = x + (size_t)rid * D_;
  float s = 0.f;
  #pragma unroll
  for (int c = tid; c < D_; c += 128) {
    float v = row[c];
    s += v * v;
  }
  #pragma unroll
  for (int off = 16; off >= 1; off >>= 1) s += __shfl_xor(s, off, 32);
  __shared__ float ws[4];
  if ((tid & 31) == 0) ws[tid >> 5] = s;
  __syncthreads();
  if (tid == 0) {
    float tot = ws[0] + ws[1] + ws[2] + ws[3];
    float inv = 1.0f / fmaxf(sqrtf(tot), 1e-12f);
    int bb = rid / T_;
    int t  = rid - bb * T_;
    if ((t & 1) == 0) inv_na[bb * N_ + (t >> 1)] = inv;
    else              inv_nb[bb * N_ + (t >> 1)] = inv;
  }
}

// ---------------------------------------------------------------------------
// Kernel 2: fused scores = a @ b^T (cosine scales applied after accumulation)
// with running row max/argmax.
// Block: 256 threads (8 waves). Block tile: 128 M-rows x all 2048 N-cols.
// A tile (128 x 1024 f16) resident in dynamic LDS for the whole block;
// B streamed in 64x128 f16 chunks. Wave w owns C rows [16w, 16w+16).
// ---------------------------------------------------------------------------
__global__ __launch_bounds__(256) void score_rowmax_kernel(
    const float* __restrict__ x,
    const float* __restrict__ inv_na,
    const float* __restrict__ inv_nb,
    float* __restrict__ max_s,
    int* __restrict__ max_i) {
  extern __shared__ __align__(16) char smem_raw[];
  _Float16* As = (_Float16*)smem_raw;        // [TILE_M][LDA]
  _Float16* Bs = As + TILE_M * LDA;          // [TILE_N][LDB]

  const int bb  = blockIdx.y;
  const int m0  = blockIdx.x * TILE_M;
  const int tid = threadIdx.x;
  const int lane = tid & 31;
  const int w    = tid >> 5;          // wave 0..7
  const int l15  = lane & 15;
  const int hi   = (lane >> 4) & 1;   // lane half-group

  const float* xb = x + (size_t)bb * T_ * D_;

  // ---- one-time load: A tile 128 rows x 1024 K, f32 -> f16, into LDS ----
  {
    const int r  = tid >> 1;                  // 0..127
    const int c0 = (tid & 1) * 512;           // 0 or 512
    const float* src = xb + (size_t)(2 * (m0 + r)) * D_ + c0;
    _Float16* dst = As + r * LDA + c0;
    #pragma unroll 4
    for (int c = 0; c < 512; c += 8) {
      v8f d = *(const v8f*)(src + c);
      *(v8h*)(dst + c) = __builtin_convertvector(d, v8h);
    }
  }
  __syncthreads();

  float rmax[8];
  int   ridx[8];
  #pragma unroll
  for (int v = 0; v < 8; ++v) { rmax[v] = -3.4e38f; ridx[v] = 0; }

  for (int n0 = 0; n0 < N_; n0 += TILE_N) {
    v8f acc[4] = {};
    // per-column cosine scale for this N tile (lane l15 = column within subtile)
    float snb[4];
    #pragma unroll
    for (int j = 0; j < 4; ++j) snb[j] = inv_nb[bb * N_ + n0 + j * 16 + l15];

    for (int kc = 0; kc < D_; kc += KCH) {
      __syncthreads();  // protect Bs against previous step's readers
      // ---- load B chunk: 64 b-rows x 128 K, f32 -> f16 ----
      {
        const int r  = tid >> 2;              // 0..63
        const int c0 = (tid & 3) * 32;        // 0/32/64/96
        const float* src = xb + (size_t)(2 * (n0 + r) + 1) * D_ + kc + c0;
        __builtin_prefetch(src + KCH, 0, 1);
        _Float16* dst = Bs + r * LDB + c0;
        #pragma unroll
        for (int c = 0; c < 32; c += 8) {
          v8f d = *(const v8f*)(src + c);
          *(v8h*)(dst + c) = __builtin_convertvector(d, v8h);
        }
      }
      __syncthreads();

      // ---- 4 K-steps of 32: preload all fragments, then chain 4 WMMAs ----
      #pragma unroll
      for (int ks = 0; ks < KCH; ks += 32) {
        // A fragment (16x32 f16): lanes 0-15: K {0..7,16..23};
        // lanes 16-31: K {8..15,24..31}
        const _Float16* arow = As + (16 * w + l15) * LDA + kc + ks;
        union { v16h v; v8h h[2]; } ua;
        ua.h[0] = *(const v8h*)(arow + 8 * hi);
        ua.h[1] = *(const v8h*)(arow + 16 + 8 * hi);
        // B fragments (32x16 f16): lane l15 = col, same K order as A
        union { v16h v; v8h h[2]; } ub[4];
        #pragma unroll
        for (int j = 0; j < 4; ++j) {
          const _Float16* brow = Bs + (j * 16 + l15) * LDB + ks + 8 * hi;
          ub[j].h[0] = *(const v8h*)brow;
          ub[j].h[1] = *(const v8h*)(brow + 16);
        }
        #pragma unroll
        for (int j = 0; j < 4; ++j) {
          acc[j] = __builtin_amdgcn_wmma_f32_16x16x32_f16(
              false, ua.v, false, ub[j].v, (short)0, acc[j], false, false);
        }
      }
    }

    // running max/argmax update (scale by inv_nb[col]; inv_na[row] is a
    // positive row constant -> applied at epilogue without changing argmax).
    // C layout: lanes 0-15: N=lane, M=v; lanes 16-31: N=lane-16, M=8+v.
    #pragma unroll
    for (int j = 0; j < 4; ++j) {
      const int ncol = n0 + j * 16 + l15;
      #pragma unroll
      for (int v = 0; v < 8; ++v) {
        float sc = acc[j][v] * snb[j];
        if (sc > rmax[v]) { rmax[v] = sc; ridx[v] = ncol; }  // strict > keeps first max
      }
    }
  }

  // reduce across the 16 lanes holding the same C rows (width-16 xor shuffles)
  #pragma unroll
  for (int off = 8; off >= 1; off >>= 1) {
    #pragma unroll
    for (int v = 0; v < 8; ++v) {
      float om = __shfl_xor(rmax[v], off, 16);
      int   oi = __shfl_xor(ridx[v], off, 16);
      if (om > rmax[v] || (om == rmax[v] && oi < ridx[v])) { rmax[v] = om; ridx[v] = oi; }
    }
  }
  if (l15 == 0) {
    const int mrow = m0 + 16 * w + 8 * hi;
    #pragma unroll
    for (int v = 0; v < 8; ++v) {
      max_s[bb * N_ + mrow + v] = rmax[v] * inv_na[bb * N_ + mrow + v];
      max_i[bb * N_ + mrow + v] = ridx[v];
    }
  }
}

// ---------------------------------------------------------------------------
// Kernel 3: per-batch bitonic argsort (desc score, asc idx tiebreak ==
// stable argsort of -scores), sequential greedy dedup, inverse row map.
// grid = B blocks, 256 threads.
// inv[t] encoding: kind<<28 | idx; kind 0=zero row, 1=a-row, 2=b-row, 3=pair s
// ---------------------------------------------------------------------------
__global__ __launch_bounds__(256) void greedy_kernel(
    const float* __restrict__ max_s,
    const int* __restrict__ max_i,
    int* __restrict__ orderArr,
    int* __restrict__ biordArr,
    int* __restrict__ inv) {
  __shared__ float         s_score[N_];
  __shared__ int           s_idx[N_];
  __shared__ int           s_bidx[N_];     // max_i in natural order
  __shared__ unsigned char s_sel[N_];
  __shared__ unsigned char s_asel[N_];
  __shared__ unsigned char s_bsel[N_];
  __shared__ unsigned int  s_used[N_ / 32];

  const int bb  = blockIdx.x;
  const int tid = threadIdx.x;

  for (int t = tid; t < T_; t += 256) inv[bb * T_ + t] = 0;
  for (int i = tid; i < N_; i += 256) {
    s_score[i] = max_s[bb * N_ + i];
    s_idx[i]   = i;
    s_bidx[i]  = max_i[bb * N_ + i];
    s_sel[i] = 0; s_asel[i] = 0; s_bsel[i] = 0;
  }
  for (int i = tid; i < N_ / 32; i += 256) s_used[i] = 0;
  __syncthreads();

  // bitonic sort: descending score, ascending original index on ties
  for (int k = 2; k <= N_; k <<= 1) {
    for (int j = k >> 1; j > 0; j >>= 1) {
      for (int i = tid; i < N_; i += 256) {
        const int ixj = i ^ j;
        if (ixj > i) {
          const bool dir_desc = ((i & k) == 0);
          float s1 = s_score[i], s2 = s_score[ixj];
          int   i1 = s_idx[i],   i2 = s_idx[ixj];
          const bool firstBetter = (s1 > s2) || (s1 == s2 && i1 < i2);
          const bool doswap = dir_desc ? !firstBetter : firstBetter;
          if (doswap) {
            s_score[i] = s2; s_score[ixj] = s1;
            s_idx[i]   = i2; s_idx[ixj]   = i1;
          }
        }
      }
      __syncthreads();
    }
  }

  // persist order / bi_ord for the scatter kernel
  for (int s = tid; s < N_; s += 256) {
    const int oi = s_idx[s];
    orderArr[bb * N_ + s] = oi;
    biordArr[bb * N_ + s] = s_bidx[oi];
  }
  __syncthreads();

  // sequential greedy matching + position assignment (thread 0)
  if (tid == 0) {
    int cnt = 0;
    for (int s = 0; s < N_; ++s) {
      const int bi = s_bidx[s_idx[s]];
      const unsigned int m = 1u << (bi & 31);
      const bool used = (s_used[bi >> 5] & m) != 0;
      const bool take = (!used) && (cnt < R_);
      s_sel[s] = (unsigned char)take;
      if (take) {
        s_used[bi >> 5] |= m;
        ++cnt;
        s_asel[s_idx[s]] = 1;
        s_bsel[bi] = 1;
      }
    }
    // output layout: [a_keep asc][pairs greedy order][b_keep asc][zeros]
    int* invb = inv + bb * T_;
    int c = 0;
    for (int i = 0; i < N_; ++i) if (!s_asel[i]) invb[c++] = (1 << 28) | i;
    for (int s = 0; s < N_; ++s) if (s_sel[s])   invb[c++] = (3 << 28) | s;
    for (int j = 0; j < N_; ++j) if (!s_bsel[j]) invb[c++] = (2 << 28) | j;
  }
}

// ---------------------------------------------------------------------------
// Kernel 4: scatter/merge rows into output. One block per output row,
// 256 threads x float4 = 1024 floats.
// ---------------------------------------------------------------------------
__global__ __launch_bounds__(256) void scatter_kernel(
    const float* __restrict__ x,
    const int* __restrict__ inv,
    const int* __restrict__ orderArr,
    const int* __restrict__ biordArr,
    float* __restrict__ out) {
  const int t   = blockIdx.x;
  const int bb  = blockIdx.y;
  const int tid = threadIdx.x;

  const int code = inv[bb * T_ + t];
  const int kind = code >> 28;
  const int idx  = code & 0x0FFFFFFF;

  float4* dst = (float4*)(out + ((size_t)bb * T_ + t) * D_);
  const float* xb = x + (size_t)bb * T_ * D_;

  if (kind == 0) {
    dst[tid] = make_float4(0.f, 0.f, 0.f, 0.f);
  } else if (kind == 1) {
    const float4* s = (const float4*)(xb + (size_t)(2 * idx) * D_);
    dst[tid] = s[tid];
  } else if (kind == 2) {
    const float4* s = (const float4*)(xb + (size_t)(2 * idx + 1) * D_);
    dst[tid] = s[tid];
  } else {
    const int ia = orderArr[bb * N_ + idx];
    const int ib = biordArr[bb * N_ + idx];
    const float4* pa = (const float4*)(xb + (size_t)(2 * ia) * D_);
    const float4* pb = (const float4*)(xb + (size_t)(2 * ib + 1) * D_);
    float4 va = pa[tid], vb = pb[tid];
    dst[tid] = make_float4(0.5f * (va.x + vb.x), 0.5f * (va.y + vb.y),
                           0.5f * (va.z + vb.z), 0.5f * (va.w + vb.w));
  }
}

// ---------------------------------------------------------------------------
extern "C" void kernel_launch(void* const* d_in, const int* in_sizes, int n_in,
                              void* d_out, int out_size, void* d_ws, size_t ws_size,
                              hipStream_t stream) {
  (void)in_sizes; (void)n_in; (void)out_size; (void)ws_size;
  const float* x = (const float*)d_in[0];
  float* out = (float*)d_out;

  // workspace layout (~512 KB total)
  float* inv_na   = (float*)d_ws;            // B*N
  float* inv_nb   = inv_na + B_ * N_;        // B*N
  float* max_s    = inv_nb + B_ * N_;        // B*N
  int*   max_i    = (int*)(max_s + B_ * N_); // B*N
  int*   orderArr = max_i + B_ * N_;         // B*N
  int*   biordArr = orderArr + B_ * N_;      // B*N
  int*   inv      = biordArr + B_ * N_;      // B*T

  // allow the 275KB dynamic-LDS allocation (CDNA5 WGP: up to 320KB)
  (void)hipFuncSetAttribute((const void*)score_rowmax_kernel,
                            hipFuncAttributeMaxDynamicSharedMemorySize,
                            (int)SMEM_BYTES);

  norm_kernel<<<dim3(B_ * T_), 128, 0, stream>>>(x, inv_na, inv_nb);
  score_rowmax_kernel<<<dim3(N_ / TILE_M, B_), 256, SMEM_BYTES, stream>>>(
      x, inv_na, inv_nb, max_s, max_i);
  greedy_kernel<<<dim3(B_), 256, 0, stream>>>(max_s, max_i, orderArr, biordArr, inv);
  scatter_kernel<<<dim3(T_, B_), 256, 0, stream>>>(x, inv, orderArr, biordArr, out);
}
